// GPLinear_23467701305624
// MI455X (gfx1250) — matrix operand order repaired and checked by reading, not compiled, not run on hardware
//
#include <hip/hip_runtime.h>

typedef _Float16 h4  __attribute__((ext_vector_type(4)));
typedef _Float16 h8  __attribute__((ext_vector_type(8)));
typedef _Float16 v16h __attribute__((ext_vector_type(16)));
typedef float    v8f __attribute__((ext_vector_type(8)));

#define GA_M    512
#define GA_N    4096
#define GA_K    4096
#define BM      128
#define BN      128
#define BK      32
#define A_STR   40    // halfs per row: 32 + 8 pad -> 80 B row stride (16B multiple)
#define B_STR   40    // halfs per (transposed) row

// out[b, o*8+k] = sum_{p,i} x[b, p*8+i] * sign(i, i^k) * W[p, o, i^k]  + bias[o*8+k]
// GEMM: A = x (M x K), B = permuted/signed W (K x N), C = out (M x N).
// Split-f16: A = Ahi+Alo, B = Bhi+Blo; acc += Ahi*Bhi + Ahi*Blo + Alo*Bhi (f32 accum).
__global__ __launch_bounds__(256)
void ga_gemm_wmma_f16x2(const float* __restrict__ X, const float* __restrict__ W,
                        const float* __restrict__ bias, float* __restrict__ out)
{
    __shared__ __align__(16) _Float16 AsHi[BM * A_STR];
    __shared__ __align__(16) _Float16 AsLo[BM * A_STR];
    __shared__ __align__(16) _Float16 BsHi[BN * B_STR];   // transposed: [n][k]
    __shared__ __align__(16) _Float16 BsLo[BN * B_STR];

    const int tid  = threadIdx.x;
    const int lane = tid & 31;
    const int wave = tid >> 5;
    const int l15  = lane & 15;
    const int half = lane >> 4;
    const int wm0  = (wave >> 2) * 64;   // 2 waves in M
    const int wn0  = (wave & 3) * 32;    // 4 waves in N
    const int bm0  = blockIdx.y * BM;
    const int bn0  = blockIdx.x * BN;

    v8f acc[4][2] = {};                  // 64x32 per wave = 4x2 fragments of 16x16

    for (int kt = 0; kt < GA_K; kt += BK) {
        // ---- stage A tile: 128x32 f32 -> hi/lo f16 planes ----
        #pragma unroll
        for (int t = 0; t < 4; ++t) {
            int idx = tid + t * 256;                 // 0..1023 float4 chunks
            int row = idx >> 3;                      // 8 float4 per row
            int c4  = (idx & 7) << 2;
            const float4 v =
                *reinterpret_cast<const float4*>(&X[(size_t)(bm0 + row) * GA_K + kt + c4]);
            h4 hi, lo;
            hi[0] = (_Float16)v.x;  lo[0] = (_Float16)(v.x - (float)hi[0]);
            hi[1] = (_Float16)v.y;  lo[1] = (_Float16)(v.y - (float)hi[1]);
            hi[2] = (_Float16)v.z;  lo[2] = (_Float16)(v.z - (float)hi[2]);
            hi[3] = (_Float16)v.w;  lo[3] = (_Float16)(v.w - (float)hi[3]);
            *reinterpret_cast<h4*>(&AsHi[row * A_STR + c4]) = hi;
            *reinterpret_cast<h4*>(&AsLo[row * A_STR + c4]) = lo;
            if (kt + BK < GA_K)
                __builtin_prefetch(&X[(size_t)(bm0 + row) * GA_K + kt + BK + c4], 0, 0);
        }
        // ---- stage B tile (transposed [n][k]): Cayley permutation + sign fused ----
        #pragma unroll
        for (int t = 0; t < 16; ++t) {
            int idx = tid + t * 256;                 // 0..4095 elements
            int kk  = idx >> 7;                      // 0..31 (K within tile)
            int n   = idx & 127;                     // 0..127 (N within tile)
            int gk  = kt + kk;
            int p   = gk >> 3, i = gk & 7;
            int gn  = bn0 + n;
            int o   = gn >> 3, kb = gn & 7;
            int j   = i ^ kb;
            int sw  = __popc((i >> 1) & j) + __popc((i >> 2) & j);
            float v = W[(((size_t)p << 9) + o) * 8 + j];
            v = (sw & 1) ? -v : v;
            _Float16 hi = (_Float16)v;
            _Float16 lo = (_Float16)(v - (float)hi);
            BsHi[n * B_STR + kk] = hi;
            BsLo[n * B_STR + kk] = lo;
            if (kt + BK < GA_K)
                __builtin_prefetch(&W[(((size_t)((gk + BK) >> 3) << 9) + o) * 8 + j], 0, 0);
        }
        __syncthreads();

        // ---- one 16x16x32 K-slice per tile iteration ----
        {
            const int kA0 = half * 8;     // A chunk0 base; chunk1 at 16 + half*8
            const int kB0 = half * 8;     // B chunk0 base; chunk1 at 16 + half*8
            v16h ah[4], al[4], bh[2], bl[2];
            #pragma unroll
            for (int mf = 0; mf < 4; ++mf) {
                const int base = (wm0 + mf * 16 + l15) * A_STR;
                h8 h0 = *reinterpret_cast<const h8*>(&AsHi[base + kA0]);
                h8 h1 = *reinterpret_cast<const h8*>(&AsHi[base + 16 + kA0]);
                ah[mf] = __builtin_shufflevector(h0, h1, 0,1,2,3,4,5,6,7,8,9,10,11,12,13,14,15);
                h8 l0 = *reinterpret_cast<const h8*>(&AsLo[base + kA0]);
                h8 l1 = *reinterpret_cast<const h8*>(&AsLo[base + 16 + kA0]);
                al[mf] = __builtin_shufflevector(l0, l1, 0,1,2,3,4,5,6,7,8,9,10,11,12,13,14,15);
            }
            #pragma unroll
            for (int nf = 0; nf < 2; ++nf) {
                const int base = (wn0 + nf * 16 + l15) * B_STR + kB0;
                h8 h0 = *reinterpret_cast<const h8*>(&BsHi[base]);
                h8 h1 = *reinterpret_cast<const h8*>(&BsHi[base + 16]);
                bh[nf] = __builtin_shufflevector(h0, h1, 0,1,2,3,4,5,6,7,8,9,10,11,12,13,14,15);
                h8 l0 = *reinterpret_cast<const h8*>(&BsLo[base]);
                h8 l1 = *reinterpret_cast<const h8*>(&BsLo[base + 16]);
                bl[nf] = __builtin_shufflevector(l0, l1, 0,1,2,3,4,5,6,7,8,9,10,11,12,13,14,15);
            }
            #pragma unroll
            for (int mf = 0; mf < 4; ++mf) {
                #pragma unroll
                for (int nf = 0; nf < 2; ++nf) {
                    acc[mf][nf] = __builtin_amdgcn_wmma_f32_16x16x32_f16(
                        false, ah[mf], false, bh[nf], (short)0, acc[mf][nf], false, false);
                    acc[mf][nf] = __builtin_amdgcn_wmma_f32_16x16x32_f16(
                        false, ah[mf], false, bl[nf], (short)0, acc[mf][nf], false, false);
                    acc[mf][nf] = __builtin_amdgcn_wmma_f32_16x16x32_f16(
                        false, al[mf], false, bh[nf], (short)0, acc[mf][nf], false, false);
                }
            }
        }
        __syncthreads();
    }

    // ---- epilogue: add bias (flat N index) and store ----
    #pragma unroll
    for (int nf = 0; nf < 2; ++nf) {
        const int n_g = bn0 + wn0 + nf * 16 + l15;
        const float bv = bias[n_g];
        #pragma unroll
        for (int mf = 0; mf < 4; ++mf) {
            const int m_base = bm0 + wm0 + mf * 16 + half * 8;
            #pragma unroll
            for (int r = 0; r < 8; ++r)
                out[(size_t)(m_base + r) * GA_N + n_g] = acc[mf][nf][r] + bv;
        }
    }
}

extern "C" void kernel_launch(void* const* d_in, const int* in_sizes, int n_in,
                              void* d_out, int out_size, void* d_ws, size_t ws_size,
                              hipStream_t stream) {
    (void)in_sizes; (void)n_in; (void)out_size; (void)d_ws; (void)ws_size;
    const float* x    = (const float*)d_in[0];   // [512, 512, 8]
    const float* W    = (const float*)d_in[1];   // [512, 512, 8]
    const float* bias = (const float*)d_in[2];   // [512, 8]
    // d_in[3] is G; the Cl(3,0) Cayley structure (xor blade, transposition sign)
    // is computed analytically in the kernel, so G itself is not needed.
    float* out = (float*)d_out;                  // [512, 512, 8]

    dim3 grid(GA_N / BN, GA_M / BM);             // (32, 4)
    ga_gemm_wmma_f16x2<<<grid, 256, 0, stream>>>(x, W, bias, out);
}
